// OffsetBlock_50603304682178
// MI455X (gfx1250) — matrix operand order repaired and checked by reading, not compiled, not run on hardware
//
#include <hip/hip_runtime.h>

// ---------------------------------------------------------------------------
// Problem constants (from reference)
// ---------------------------------------------------------------------------
#define B_        16
#define CIN_      256
#define COUT_     256
#define DP_       512
#define H_        64
#define W_        64
#define HW_       4096            // H_*W_
#define EPS_      1.1920928955078125e-07f
#define BN_EPS_   1e-05f
#define CLAMP_MAX_ 88.722835f

// CDNA5 WMMA operand types (wave32)
typedef __attribute__((ext_vector_type(16))) __bf16 bf16x16;
typedef __attribute__((ext_vector_type(8)))  __bf16 bf16x8;
typedef __attribute__((ext_vector_type(8)))  float  f32x8;
typedef __attribute__((ext_vector_type(4)))  float  f32x4;

// LDS tile row pitch in bf16 elements: 32 used + 8 pad.
// 80-byte row stride => 16B-aligned b128 chunks, 20-bank stride => the 16 lanes
// of a half-wave cover all 64 LDS banks (conflict-free ds_load_b128 gathers).
#define APITCH_ 40

// pack 4 floats -> 4 bf16, single ds_store_b64
__device__ __forceinline__ void store4bf16(__bf16* dst, f32x4 v) {
    union { __bf16 h[4]; unsigned long long q; } t;
    t.h[0] = (__bf16)v[0]; t.h[1] = (__bf16)v[1];
    t.h[2] = (__bf16)v[2]; t.h[3] = (__bf16)v[3];
    *(unsigned long long*)dst = t.q;
}

__device__ __forceinline__ bf16x16 cat8(bf16x8 lo, bf16x8 hi) {
    return __builtin_shufflevector(lo, hi, 0,1,2,3,4,5,6,7,8,9,10,11,12,13,14,15);
}

// ---------------------------------------------------------------------------
// Stage A: fused dual GEMM.
//   pre[b,dp,hw] = sum_c x[b,c,hw]*w_pre[dp,c] + b_pre[dp] -> exp(min(.,CLAMP))
//   att[b,dp,hw] = sum_c x[b,c,hw]*w_att[dp,c] + b_att[dp]  (raw + BN stats)
// Block tile M(dp)=64 x N(pixels)=64; 8 waves, each 16x32; K step 32.
// x tile staged once in LDS (transposed [n][k]) and shared by both GEMMs.
// ---------------------------------------------------------------------------
__global__ __launch_bounds__(256)
void gemm_pre_att_kernel(const float* __restrict__ x,
                         const float* __restrict__ w_pre, const float* __restrict__ b_pre,
                         const float* __restrict__ w_att, const float* __restrict__ b_att,
                         float* __restrict__ pre_exp, float* __restrict__ att_lin,
                         float* __restrict__ attSum, float* __restrict__ attSumSq)
{
    __shared__ __align__(16) __bf16 lA[2][64][APITCH_];   // [m][k] row-major
    __shared__ __align__(16) __bf16 lBt[64][APITCH_];     // [n][k] transposed

    const int tid   = threadIdx.x;
    const int lane  = tid & 31;
    const int wave  = tid >> 5;
    const int waveM = wave >> 1;              // 0..3
    const int waveN = wave & 1;               // 0..1
    const int Mbase = blockIdx.y * 64;        // dp
    const int Nbase = blockIdx.x * 64;        // flattened (b,hw); 64 cols stay in 1 batch
    const int bN     = Nbase >> 12;
    const int hwBase = Nbase & (HW_ - 1);

    const int m = lane & 15;
    const int g = lane >> 4;

    f32x8 accP[2] = {};
    f32x8 accA[2] = {};

    for (int kk = 0; kk < CIN_; kk += 32) {
        // ---- stage tiles: vector global loads, bulk convert, packed LDS stores
        const int rA = tid >> 3;              // 0..31
        const int kc = (tid & 7) * 4;         // 0..28
        f32x4 wp0 = *(const f32x4*)&w_pre[(Mbase + rA     ) * CIN_ + kk + kc];
        f32x4 wp1 = *(const f32x4*)&w_pre[(Mbase + rA + 32) * CIN_ + kk + kc];
        f32x4 wa0 = *(const f32x4*)&w_att[(Mbase + rA     ) * CIN_ + kk + kc];
        f32x4 wa1 = *(const f32x4*)&w_att[(Mbase + rA + 32) * CIN_ + kk + kc];
        const int krB = tid >> 4;             // 0..15
        const int cB  = (tid & 15) * 4;       // 0..60
        f32x4 xb0 = *(const f32x4*)&x[((size_t)(bN*CIN_ + kk + krB     ) << 12) + hwBase + cB];
        f32x4 xb1 = *(const f32x4*)&x[((size_t)(bN*CIN_ + kk + krB + 16) << 12) + hwBase + cB];

        store4bf16(&lA[0][rA     ][kc], wp0);
        store4bf16(&lA[0][rA + 32][kc], wp1);
        store4bf16(&lA[1][rA     ][kc], wa0);
        store4bf16(&lA[1][rA + 32][kc], wa1);
        #pragma unroll
        for (int j = 0; j < 4; ++j) {         // transpose on store
            lBt[cB + j][krB     ] = (__bf16)xb0[j];
            lBt[cB + j][krB + 16] = (__bf16)xb1[j];
        }
        if (kk + 32 < CIN_) {                 // global_prefetch_b8 of next k-tile
            __builtin_prefetch(&x[((size_t)(bN*CIN_ + kk + 32 + krB) << 12) + hwBase + cB], 0, 0);
        }
        __syncthreads();

        // ---- fragments: two ds_load_b128 each (CDNA5 wave32 WMMA layouts)
        const __bf16* ar = &lA[0][waveM*16 + m][0];
        bf16x16 aP = cat8(*(const bf16x8*)(ar + g*8),
                          *(const bf16x8*)(ar + 16 + g*8));
        const __bf16* ar2 = &lA[1][waveM*16 + m][0];
        bf16x16 aA = cat8(*(const bf16x8*)(ar2 + g*8),
                          *(const bf16x8*)(ar2 + 16 + g*8));
        #pragma unroll
        for (int sub = 0; sub < 2; ++sub) {
            const __bf16* br = &lBt[waveN*32 + sub*16 + m][g*8];
            bf16x16 bF = cat8(*(const bf16x8*)br, *(const bf16x8*)(br + 16));
            accP[sub] = __builtin_amdgcn_wmma_f32_16x16x32_bf16(
                false, aP, false, bF, (short)0, accP[sub], false, false);
            accA[sub] = __builtin_amdgcn_wmma_f32_16x16x32_bf16(
                false, aA, false, bF, (short)0, accA[sub], false, false);
        }
        __syncthreads();
    }

    // ---- epilogue: bias + exp/clamp for pre; raw store + BN stats for att
    #pragma unroll
    for (int sub = 0; sub < 2; ++sub) {
        #pragma unroll
        for (int j = 0; j < 8; ++j) {
            int dp = Mbase + waveM * 16 + (g * 8 + j);
            int hw = hwBase + waveN * 32 + sub * 16 + m;
            size_t oidx = ((size_t)(bN * DP_ + dp) << 12) + hw;

            float vp = accP[sub][j] + b_pre[dp];
            pre_exp[oidx] = expf(fminf(vp, CLAMP_MAX_));

            float va = accA[sub][j] + b_att[dp];
            att_lin[oidx] = va;

            float s = va, s2 = va * va;     // lanes sharing this M-row = low-4-bit group
            #pragma unroll
            for (int off = 1; off < 16; off <<= 1) {
                s  += __shfl_xor(s,  off, 32);
                s2 += __shfl_xor(s2, off, 32);
            }
            if (m == 0) {
                atomicAdd(&attSum[dp],   s);
                atomicAdd(&attSumSq[dp], s2);
            }
        }
    }
}

// ---------------------------------------------------------------------------
// Stage B: dpool. Per (b,c) image: gp = conv5x5(pre_exp, gauss(sigma_c)),
// dlog = log(gp + EPS). Gaussian taps + 8x64 halo tile built in LDS.
// ---------------------------------------------------------------------------
__global__ __launch_bounds__(256)
void dpool_kernel(const float* __restrict__ pre_exp, const float* __restrict__ sigma,
                  float* __restrict__ dlog)
{
    __shared__ float lw[26];                  // 25 taps + [25] = sum
    __shared__ float limg[8][64];             // rows row0-2 .. row0+5 (zero padded)
    const int tid  = threadIdx.x;
    const int bc   = blockIdx.x;              // b*DP + c
    const int c    = bc & (DP_ - 1);
    const int row0 = blockIdx.y * 4;          // 4 output rows per block
    const float* img = pre_exp + ((size_t)bc << 12);

    if (tid < 25) {
        int dy = tid / 5 - 2, dx = tid % 5 - 2;
        float sg = sigma[c];
        lw[tid] = __expf(-0.5f * (float)(dy * dy + dx * dx) / (sg * sg + EPS_));
    }
    #pragma unroll
    for (int i = 0; i < 2; ++i) {             // halo load, 2 elements/thread
        int e  = tid + i * 256;
        int ry = e >> 6, xs = e & 63;
        int gy = row0 - 2 + ry;
        limg[ry][xs] = (gy >= 0 && gy < H_) ? img[(gy << 6) + xs] : 0.0f;
    }
    __syncthreads();
    if (tid == 0) {
        float s = 0.f;
        #pragma unroll
        for (int i = 0; i < 25; ++i) s += lw[i];
        lw[25] = s;
    }
    __syncthreads();

    const float inv = 1.0f / lw[25];
    const int yl = tid >> 6;                  // 0..3
    const int xx = tid & 63;
    float acc = 0.f;
    #pragma unroll
    for (int dy = -2; dy <= 2; ++dy) {
        const float* lrow = limg[yl + 2 + dy];
        #pragma unroll
        for (int dx = -2; dx <= 2; ++dx) {
            int xs = xx + dx;
            if (xs < 0 || xs >= W_) continue;
            acc += lw[(dy + 2) * 5 + (dx + 2)] * lrow[xs];
        }
    }
    dlog[((size_t)bc << 12) + ((row0 + yl) << 6) + xx] = logf(acc * inv + EPS_);
}

// ---------------------------------------------------------------------------
// BN statistics finalize: mean = sum/N, rstd = rsqrt(var + eps) (population)
// ---------------------------------------------------------------------------
__global__ void bn_finalize_kernel(const float* __restrict__ sum, const float* __restrict__ sumsq,
                                   float* __restrict__ mean, float* __restrict__ rstd,
                                   int nchan, float invN)
{
    int c = blockIdx.x * blockDim.x + threadIdx.x;
    if (c >= nchan) return;
    float mn = sum[c] * invN;
    float v  = sumsq[c] * invN - mn * mn;
    mean[c] = mn;
    rstd[c] = rsqrtf(v + BN_EPS_);
}

// ---------------------------------------------------------------------------
// Stage D: displace (bilinear shift, zero border) * sigmoid(BN(att_lin)),
// written as bf16 — directly consumable as WMMA B-operand by stage E.
// ---------------------------------------------------------------------------
__global__ __launch_bounds__(256)
void displace_att_kernel(const float* __restrict__ dlog, const float* __restrict__ att_lin,
                         const float* __restrict__ offsets,
                         const float* __restrict__ g_att, const float* __restrict__ be_att,
                         const float* __restrict__ mean, const float* __restrict__ rstd,
                         __bf16* __restrict__ mid)
{
    const size_t idx = (size_t)blockIdx.x * 256 + threadIdx.x;
    const int p = (int)(idx & (HW_ - 1));
    const int c = (int)((idx >> 12) & (DP_ - 1));
    const int y = p >> 6, xx = p & 63;

    const float ox = offsets[c * 2 + 0], oy = offsets[c * 2 + 1];
    float sy = (float)y + oy, sx = (float)xx + ox;
    float y0f = floorf(sy), x0f = floorf(sx);
    float fy = sy - y0f,    fx = sx - x0f;
    int   y0 = (int)y0f,    x0 = (int)x0f;

    const float* img = dlog + (idx & ~(size_t)(HW_ - 1));
    auto G = [&](int iy, int ix) -> float {
        if (iy < 0 || iy >= H_ || ix < 0 || ix >= W_) return 0.0f;
        return img[(iy << 6) + ix];
    };
    float v = G(y0,     x0    ) * (1.f - fy) * (1.f - fx)
            + G(y0,     x0 + 1) * (1.f - fy) * fx
            + G(y0 + 1, x0    ) * fy * (1.f - fx)
            + G(y0 + 1, x0 + 1) * fy * fx;

    float t   = (att_lin[idx] - mean[c]) * rstd[c] * g_att[c] + be_att[c];
    float att = 1.0f / (1.0f + expf(-t));
    mid[idx]  = (__bf16)(v * att);
}

// ---------------------------------------------------------------------------
// Stage E: GEMM3 + residual + output-BN stats.
//   y[b,o,hw] = x[b,o,hw] + b_post[o] + sum_dp w_post[o,dp] * mid[b,dp,hw]
// ---------------------------------------------------------------------------
__global__ __launch_bounds__(256)
void gemm_post_kernel(const __bf16* __restrict__ mid, const float* __restrict__ w_post,
                      const float* __restrict__ b_post, const float* __restrict__ x,
                      float* __restrict__ ybuf,
                      float* __restrict__ outSum, float* __restrict__ outSumSq)
{
    __shared__ __align__(16) __bf16 lA[64][APITCH_];
    __shared__ __align__(16) __bf16 lBt[64][APITCH_];

    const int tid   = threadIdx.x;
    const int lane  = tid & 31;
    const int wave  = tid >> 5;
    const int waveM = wave >> 1;
    const int waveN = wave & 1;
    const int Mbase = blockIdx.y * 64;        // cout
    const int Nbase = blockIdx.x * 64;
    const int bN     = Nbase >> 12;
    const int hwBase = Nbase & (HW_ - 1);

    const int m = lane & 15;
    const int g = lane >> 4;

    f32x8 acc[2] = {};

    for (int kk = 0; kk < DP_; kk += 32) {
        const int rA = tid >> 3;
        const int kc = (tid & 7) * 4;
        f32x4 wp0 = *(const f32x4*)&w_post[(Mbase + rA     ) * DP_ + kk + kc];
        f32x4 wp1 = *(const f32x4*)&w_post[(Mbase + rA + 32) * DP_ + kk + kc];
        const int krB = tid >> 3;             // 0..31
        const int cB  = (tid & 7) * 8;        // 0..56
        bf16x8 mb = *(const bf16x8*)&mid[((size_t)(bN*DP_ + kk + krB) << 12) + hwBase + cB];

        store4bf16(&lA[rA     ][kc], wp0);
        store4bf16(&lA[rA + 32][kc], wp1);
        #pragma unroll
        for (int j = 0; j < 8; ++j)           // transpose on store
            lBt[cB + j][krB] = mb[j];
        if (kk + 32 < DP_) {
            __builtin_prefetch(&mid[((size_t)(bN*DP_ + kk + 32 + krB) << 12) + hwBase + cB], 0, 0);
        }
        __syncthreads();

        const __bf16* ar = &lA[waveM*16 + m][0];
        bf16x16 aF = cat8(*(const bf16x8*)(ar + g*8),
                          *(const bf16x8*)(ar + 16 + g*8));
        #pragma unroll
        for (int sub = 0; sub < 2; ++sub) {
            const __bf16* br = &lBt[waveN*32 + sub*16 + m][g*8];
            bf16x16 bF = cat8(*(const bf16x8*)br, *(const bf16x8*)(br + 16));
            acc[sub] = __builtin_amdgcn_wmma_f32_16x16x32_bf16(
                false, aF, false, bF, (short)0, acc[sub], false, false);
        }
        __syncthreads();
    }

    #pragma unroll
    for (int sub = 0; sub < 2; ++sub) {
        #pragma unroll
        for (int j = 0; j < 8; ++j) {
            int o  = Mbase + waveM * 16 + (g * 8 + j);
            int hw = hwBase + waveN * 32 + sub * 16 + m;
            size_t oidx = ((size_t)(bN * COUT_ + o) << 12) + hw;

            float yv = acc[sub][j] + b_post[o] + x[oidx];   // residual (CIN == COUT)
            ybuf[oidx] = yv;

            float s = yv, s2 = yv * yv;
            #pragma unroll
            for (int off = 1; off < 16; off <<= 1) {
                s  += __shfl_xor(s,  off, 32);
                s2 += __shfl_xor(s2, off, 32);
            }
            if (m == 0) {
                atomicAdd(&outSum[o],   s);
                atomicAdd(&outSumSq[o], s2);
            }
        }
    }
}

// ---------------------------------------------------------------------------
// Stage G: final BN + ReLU writeout (fp32 output)
// ---------------------------------------------------------------------------
__global__ __launch_bounds__(256)
void bn_relu_kernel(const float* __restrict__ ybuf, const float* __restrict__ mean,
                    const float* __restrict__ rstd, const float* __restrict__ g,
                    const float* __restrict__ be, float* __restrict__ out)
{
    size_t idx = (size_t)blockIdx.x * 256 + threadIdx.x;
    int c = (int)((idx >> 12) & (COUT_ - 1));
    float t = (ybuf[idx] - mean[c]) * rstd[c] * g[c] + be[c];
    out[idx] = fmaxf(t, 0.0f);
}

// ---------------------------------------------------------------------------
// Host-side launcher (graph-capture safe: async ops on `stream` only)
// ---------------------------------------------------------------------------
extern "C" void kernel_launch(void* const* d_in, const int* in_sizes, int n_in,
                              void* d_out, int out_size, void* d_ws, size_t ws_size,
                              hipStream_t stream)
{
    const float* x       = (const float*)d_in[0];
    const float* w_pre   = (const float*)d_in[1];
    const float* b_pre   = (const float*)d_in[2];
    const float* sigma   = (const float*)d_in[3];
    const float* offsets = (const float*)d_in[4];
    const float* w_att   = (const float*)d_in[5];
    const float* b_att   = (const float*)d_in[6];
    const float* g_att   = (const float*)d_in[7];
    const float* be_att  = (const float*)d_in[8];
    const float* w_post  = (const float*)d_in[9];
    const float* b_post  = (const float*)d_in[10];
    const float* g_bn    = (const float*)d_in[11];
    const float* be_bn   = (const float*)d_in[12];

    char* ws = (char*)d_ws;
    const size_t SZ_BDHW = (size_t)B_ * DP_ * HW_ * sizeof(float);   // 128 MiB

    float*  pre_exp = (float*)(ws);                   // [0, 128M)
    float*  att_lin = (float*)(ws + SZ_BDHW);         // [128M, 256M)
    float*  dlog    = (float*)(ws + 2 * SZ_BDHW);     // [256M, 384M)
    float*  stats   = (float*)(ws + 3 * SZ_BDHW);     // 12 KB of stats
    __bf16* mid     = (__bf16*)(ws);                  // reuse pre_exp (dead after dpool)
    float*  ybuf    = (float*)(ws + SZ_BDHW);         // reuse att_lin (dead after stage D)

    float* attSum   = stats;
    float* attSumSq = stats + 512;
    float* attMean  = stats + 1024;
    float* attRstd  = stats + 1536;
    float* outSum   = stats + 2048;
    float* outSumSq = stats + 2304;
    float* outMean  = stats + 2560;
    float* outRstd  = stats + 2816;

    hipMemsetAsync(stats, 0, 3072 * sizeof(float), stream);

    dim3 blk(256);
    // Stage A: fused dual WMMA GEMM (pre + att), x tile shared in LDS
    gemm_pre_att_kernel<<<dim3((B_ * HW_) / 64, DP_ / 64), blk, 0, stream>>>(
        x, w_pre, b_pre, w_att, b_att, pre_exp, att_lin, attSum, attSumSq);
    // Stage B: depthwise Gaussian pool in exp-domain
    dpool_kernel<<<dim3(B_ * DP_, HW_ / 256), blk, 0, stream>>>(pre_exp, sigma, dlog);
    // Stage C: att BN stats
    bn_finalize_kernel<<<dim3(2), blk, 0, stream>>>(attSum, attSumSq, attMean, attRstd,
                                                    DP_, 1.0f / (float)(B_ * HW_));
    // Stage D: displace * sigmoid(BN(att)) -> bf16 mid
    displace_att_kernel<<<dim3((B_ * DP_ * HW_) / 256), blk, 0, stream>>>(
        dlog, att_lin, offsets, g_att, be_att, attMean, attRstd, mid);
    // Stage E: WMMA GEMM3 + residual + output BN stats
    gemm_post_kernel<<<dim3((B_ * HW_) / 64, COUT_ / 64), blk, 0, stream>>>(
        mid, w_post, b_post, x, ybuf, outSum, outSumSq);
    // Stage F: output BN stats
    bn_finalize_kernel<<<dim3(1), blk, 0, stream>>>(outSum, outSumSq, outMean, outRstd,
                                                    COUT_, 1.0f / (float)(B_ * HW_));
    // Stage G: BN + ReLU
    bn_relu_kernel<<<dim3((B_ * COUT_ * HW_) / 256), blk, 0, stream>>>(
        ybuf, outMean, outRstd, g_bn, be_bn, (float*)d_out);
}
